// FloatingRegionScore_4501125726850
// MI455X (gfx1250) — 8-bit matrix operand order repaired and checked by reading, compiled, not run on hardware
//
#include <hip/hip_runtime.h>
#include <hip/hip_bf16.h>
#include <stdint.h>

#define H   1024
#define W   2048
#define NC  19
#define HW  (H * W)
#define SEG 64

#define EPS 1e-6f
#define INV_LOG19 (1.0f / 2.9444389791664403f)

typedef __attribute__((ext_vector_type(8))) int v8i;

__global__ __launch_bounds__(256) void k_pixel(const float* __restrict__ logit,
                                               float* __restrict__ ent,
                                               uint8_t* __restrict__ pred) {
    int idx = blockIdx.x * 256 + threadIdx.x;
    if (idx >= HW) return;
    float x[NC];
#pragma unroll
    for (int c = 0; c < NC; ++c) x[c] = logit[(size_t)c * HW + idx];
    float m = x[0]; int am = 0;
#pragma unroll
    for (int c = 1; c < NC; ++c) { if (x[c] > m) { m = x[c]; am = c; } }
    float s = 0.f;
#pragma unroll
    for (int c = 0; c < NC; ++c) { x[c] = __expf(x[c] - m); s += x[c]; }
    float inv = 1.0f / s;
    float e = 0.f;
#pragma unroll
    for (int c = 0; c < NC; ++c) { float p = x[c] * inv; e -= p * __logf(p + EPS); }
    ent[idx]  = e * INV_LOG19;
    pred[idx] = (uint8_t)am;
}

__device__ __forceinline__ void bumpCnt(uint32_t cnt[5], int c, bool add) {
    uint32_t d = 1u << ((c & 3) * 8);
    int j = c >> 2;
#pragma unroll
    for (int q = 0; q < 5; ++q) {
        uint32_t dd = (q == j) ? d : 0u;
        cnt[q] = add ? (cnt[q] + dd) : (cnt[q] - dd);
    }
}

__global__ __launch_bounds__(256) void k_vert(const float* __restrict__ ent,
                                              const uint8_t* __restrict__ pred,
                                              float* __restrict__ colEnt,
                                              uint8_t* __restrict__ cnt8) {
    int w  = blockIdx.x * 256 + threadIdx.x;
    int h0 = blockIdx.y * SEG;
    if (w >= W) return;

    uint32_t cnt[5] = {0, 0, 0, 0, 0};
    float es = 0.f;
    for (int r = max(h0 - 16, 0); r < min(h0 + 16, H); ++r) {
        bumpCnt(cnt, (int)pred[(size_t)r * W + w], true);
        es += ent[(size_t)r * W + w];
    }
    int hEnd = min(h0 + SEG, H);
    for (int h = h0; h < hEnd; ++h) {
        int rin = h + 16;
        if (rin < H) {
            bumpCnt(cnt, (int)pred[(size_t)rin * W + w], true);
            es += ent[(size_t)rin * W + w];
            int rpf = min(rin + 16, H - 1);
            __builtin_prefetch((const void*)(ent  + (size_t)rpf * W + w), 0, 0);
            __builtin_prefetch((const void*)(pred + (size_t)rpf * W + w), 0, 0);
        }
        colEnt[(size_t)h * W + w] = es;
#pragma unroll
        for (int j = 0; j < 5; ++j) {
            uint32_t v = cnt[j];
#pragma unroll
            for (int b = 0; b < 4; ++b) {
                int c = j * 4 + b;
                if (c < NC)
                    cnt8[(size_t)c * HW + (size_t)h * W + w] =
                        (uint8_t)((v >> (8 * b)) & 0xFFu);
            }
        }
        int rout = h - 16;
        if (rout >= 0) {
            bumpCnt(cnt, (int)pred[(size_t)rout * W + w], false);
            es -= ent[(size_t)rout * W + w];
        }
    }
}

__device__ __forceinline__ void loadBpair(const uint8_t* __restrict__ cnt8,
                                          int h, int w0, int n, int kgrp,
                                          v8i& B0, v8i& B1) {
    const uint8_t* base0 = cnt8 + (size_t)n * HW + (size_t)h * W;
    int n2c = (16 + n < NC) ? (16 + n) : (NC - 1);
    const uint8_t* base1 = cnt8 + (size_t)n2c * HW + (size_t)h * W;
    bool nv1 = (16 + n) < NC;
#pragma unroll
    for (int v = 0; v < 8; ++v) {
        int k0 = (v >> 2) * 32 + kgrp * 16 + (v & 3) * 4;
        int gw = w0 - 16 + k0;
        bool inb = (gw >= 0) && (gw <= W - 4);
        int gwc = min(max(gw, 0), W - 4);
        uint32_t r0 = *(const uint32_t*)(base0 + gwc);
        uint32_t r1 = *(const uint32_t*)(base1 + gwc);
        B0[v] = inb ? (int)r0 : 0;
        B1[v] = (inb && nv1) ? (int)r1 : 0;
    }
}

__global__ __launch_bounds__(32) void k_horiz(const float* __restrict__ colEnt,
                                              const uint8_t* __restrict__ cnt8,
                                              float* __restrict__ out) {
    __shared__ float sEnt[16][48];
    __shared__ int   sCnt[2][16][32];

    const int lane = threadIdx.x;
    const int w0 = blockIdx.x * 16;
    const int h0 = blockIdx.y * 16;

    for (int i = lane; i < 16 * 48; i += 32) {
        int hh = i / 48, ww = i % 48;
        int gw = w0 - 16 + ww;
        sEnt[hh][ww] = (gw >= 0 && gw < W) ? colEnt[(size_t)(h0 + hh) * W + gw] : 0.f;
    }
    __syncthreads();

    v8i A;
    {
        int m = lane & 15, hi = lane >> 4;
#pragma unroll
        for (int v = 0; v < 8; ++v) {
            int kb = (v >> 2) * 32 + hi * 16 + (v & 3) * 4;
            uint32_t word = 0;
#pragma unroll
            for (int b = 0; b < 4; ++b) {
                int k = kb + b;
                if (k >= m && k <= m + 32) word |= (1u << (8 * b));
            }
            A[v] = (int)word;
        }
    }

    const int n    = lane & 15;
    const int kgrp = lane >> 4;

    v8i cB[4], nB[4];
    loadBpair(cnt8, h0,     w0, n, kgrp, cB[0], cB[1]);
    loadBpair(cnt8, h0 + 1, w0, n, kgrp, cB[2], cB[3]);

    for (int hh = 0; hh < 16; hh += 2) {
        if (hh + 2 < 16) {
            loadBpair(cnt8, h0 + hh + 2, w0, n, kgrp, nB[0], nB[1]);
            loadBpair(cnt8, h0 + hh + 3, w0, n, kgrp, nB[2], nB[3]);
        }
        v8i Z = {};
        v8i D0 = __builtin_amdgcn_wmma_i32_16x16x64_iu8(false, A, false, cB[0], Z, false, false);
        v8i D1 = __builtin_amdgcn_wmma_i32_16x16x64_iu8(false, A, false, cB[1], Z, false, false);
        v8i D2 = __builtin_amdgcn_wmma_i32_16x16x64_iu8(false, A, false, cB[2], Z, false, false);
        v8i D3 = __builtin_amdgcn_wmma_i32_16x16x64_iu8(false, A, false, cB[3], Z, false, false);

#pragma unroll
        for (int r = 0; r < 8; ++r) {
            int M = kgrp * 8 + r;
            sCnt[0][M][n]      = D0[r];
            sCnt[0][M][16 + n] = D1[r];
            sCnt[1][M][n]      = D2[r];
            sCnt[1][M][16 + n] = D3[r];
        }
        __syncthreads();

        {
            int row = lane >> 4;
            int m   = lane & 15;
            float tot = 0.f;
#pragma unroll
            for (int c = 0; c < NC; ++c) tot += (float)sCnt[row][m][c];
            float itot = 1.0f / tot;
            float imp = 0.f;
#pragma unroll
            for (int c = 0; c < NC; ++c) {
                float d = (float)sCnt[row][m][c] * itot;
                imp -= d * __logf(d + EPS);
            }
            imp *= INV_LOG19;
            float es = 0.f;
#pragma unroll
            for (int j = 0; j <= 32; ++j) es += sEnt[hh + row][m + j];
            float unc = es * itot;
            size_t o = (size_t)(h0 + hh + row) * W + (size_t)(w0 + m);
            out[o]                  = imp * unc;
            out[(size_t)HW + o]     = imp;
            out[2 * (size_t)HW + o] = unc;
        }
        __syncthreads();

#pragma unroll
        for (int q = 0; q < 4; ++q) cB[q] = nB[q];
    }
}

extern "C" void kernel_launch(void* const* d_in, const int* in_sizes, int n_in,
                              void* d_out, int out_size, void* d_ws, size_t ws_size,
                              hipStream_t stream) {
    (void)in_sizes; (void)n_in; (void)out_size; (void)ws_size;
    const float* logit = (const float*)d_in[0];

    float*   ent    = (float*)d_ws;
    float*   colEnt = ent + (size_t)HW;
    uint8_t* pred   = (uint8_t*)(colEnt + (size_t)HW);
    uint8_t* cnt8   = pred + (size_t)HW;
    float*   out    = (float*)d_out;

    k_pixel<<<dim3(HW / 256), dim3(256), 0, stream>>>(logit, ent, pred);
    k_vert <<<dim3(W / 256, H / SEG), dim3(256), 0, stream>>>(ent, pred, colEnt, cnt8);
    k_horiz<<<dim3(W / 16, H / 16), dim3(32), 0, stream>>>(colEnt, cnt8, out);
}
